// TaskScoreModel_8272107012332
// MI455X (gfx1250) — matrix operand order repaired and checked by reading, not compiled, not run on hardware
//
#include <hip/hip_runtime.h>
#include <hip/hip_bf16.h>

// ---------------------------------------------------------------------------
// PointNet++ MSG forward for gfx1250 (MI455X).
// MLP layers = wave32 WMMA f16 GEMMs (v_wmma_f32_16x16x32_f16, f32 accum).
// Weight tiles are staged into LDS with the Tensor Data Mover
// (tensor_load_to_lds + s_wait_tensorcnt) and shared by all 8 waves of a block.
// Irregular ops (FPS / ball query / 3-NN) stay scalar: latency-, not
// throughput-bound.
// ---------------------------------------------------------------------------

typedef __attribute__((ext_vector_type(16))) _Float16      v16h;
typedef __attribute__((ext_vector_type(8)))  float         v8f;
typedef __attribute__((ext_vector_type(4)))  unsigned int  v4u;
typedef __attribute__((ext_vector_type(8)))  int           v8i;
typedef __attribute__((ext_vector_type(4)))  int           v4i;

#define BNS_C 0.9999950000374997f
#define MAX_KPAD 1664   // largest K across all layers (FP3 input)

union Frag {
  v16h  v;
  uint4 q[2];
};

// ---------------------------------------------------------------------------
// WMMA GEMM: Y[rows, Nc](f16) = act( BNS * (X[rows, Kpad](f16) @ W[Nc, Kpad]^T + b) )
// grid = (Mtiles/8, Ntiles); block = 256 (8 waves); wave w -> M-tile blockIdx.x*8+w.
// The 16 x Kpad weight strip for this N-tile is DMA'd into LDS once (TDM) and
// shared by all waves. rows % 128 == 0, Kpad % 32 == 0, Nc % 16 == 0.
// ---------------------------------------------------------------------------
__global__ __launch_bounds__(256) void k_gemm_relu(
    const _Float16* __restrict__ X,
    const _Float16* __restrict__ W,     // [Nc, Kpad], zero-padded in K
    const float*    __restrict__ bias,  // [Nc]
    _Float16*       __restrict__ Y,     // [rows, Nc]
    int Mtiles, int Kpad, int Npad, int doRelu)
{
  __shared__ __align__(16) _Float16 lw[16 * MAX_KPAD];

  const int wave = threadIdx.x >> 5;
  const int lane = threadIdx.x & 31;
  const int mt   = blockIdx.x * 8 + wave;
  const int n0   = blockIdx.y << 4;
  const int hh   = lane >> 4;    // 0/1 lane half
  const int l15  = lane & 15;

  // --- stage W[n0..n0+15, 0..Kpad) into LDS via the Tensor Data Mover ---
  if (threadIdx.x < 32) {                      // wave 0 only (uniform branch)
    const unsigned ldsOff = (unsigned)(uintptr_t)(&lw[0]);   // LDS byte offset
    const unsigned long long ga =
        (unsigned long long)(uintptr_t)(W + (size_t)n0 * Kpad);
    const unsigned td0 = (unsigned)(Kpad >> 2);  // row length in 8-byte units
    v4u g0;
    g0.x = 1u;                                          // count=1, user desc
    g0.y = ldsOff;                                      // lds_addr
    g0.z = (unsigned)(ga & 0xFFFFFFFFu);                // global_addr[31:0]
    g0.w = (unsigned)((ga >> 32) & 0x1FFFFFFu)          // global_addr[56:32]
         | (2u << 30);                                  // type = 2 ("image")
    v8i g1;
    g1[0] = (int)(3u << 16);                            // data_size=3 (8B), mask=0
    g1[1] = (int)((td0 & 0xFFFFu) << 16);               // tensor_dim0[15:0]
    g1[2] = (int)((td0 >> 16) | (16u << 16));           // tensor_dim0[31:16] | tensor_dim1=16
    g1[3] = (int)((td0 & 0xFFFFu) << 16);               // tile_dim0
    g1[4] = 16;                                         // tile_dim1=16, tile_dim2=0
    g1[5] = (int)td0;                                   // tensor_dim0_stride[31:0]
    g1[6] = 0;                                          // stride0 hi | stride1 lo
    g1[7] = 0;                                          // stride1 hi
    const v4i gz4 = {0, 0, 0, 0};
    const v8i gz8 = {0, 0, 0, 0, 0, 0, 0, 0};
    __builtin_amdgcn_tensor_load_to_lds(g0, g1, gz4, gz4, gz8, 0);
    __builtin_amdgcn_s_wait_tensorcnt(0);               // TENSORcnt per-wave
  }
  __syncthreads();

  if (mt < Mtiles) {   // uniform per wave; never taken in practice (rows%128==0)
    const int m0 = mt << 4;
    const _Float16* xrow = X + (size_t)(m0 + l15) * Kpad;
    const _Float16* brow = &lw[(size_t)l15 * Kpad];

    v8f acc = {};
    for (int k = 0; k < Kpad; k += 32) {
      Frag a, b;
      // A 16x32 f16: lanes 0-15 hold K {0..7,16..23}+base, lanes 16-31 K {8..15,24..31}
      a.q[0] = *(const uint4*)(xrow + k + 8 * hh);
      a.q[1] = *(const uint4*)(xrow + k + 16 + 8 * hh);
      // B 32x16 f16 from LDS: lanes 0-15 hold K {0..7,16..23}, lanes 16-31 K {8..15,24..31} (col = lane&15)
      b.q[0] = *(const uint4*)(brow + k + 8 * hh);
      b.q[1] = *(const uint4*)(brow + k + 16 + 8 * hh);
      if (k + 32 < Kpad) {
        __builtin_prefetch(xrow + k + 32, 0, 3);   // global_prefetch_b8
      }
      acc = __builtin_amdgcn_wmma_f32_16x16x32_f16(
          /*neg_a=*/false, a.v, /*neg_b=*/false, b.v,
          /*c_mod=*/(short)0, acc, /*reuse_a=*/false, /*reuse_b=*/false);
    }

    const float bv = bias[n0 + l15];  // this lane's output column
#pragma unroll
    for (int r = 0; r < 8; ++r) {
      float v = BNS_C * (acc[r] + bv);
      if (doRelu) v = v > 0.f ? v : 0.f;
      Y[(size_t)(m0 + r + 8 * hh) * Npad + n0 + l15] = (_Float16)v;
    }
  }
}

// ---------------------------------------------------------------------------
// Farthest point sampling: one block per batch. dist is [B,N] f32 workspace.
// ---------------------------------------------------------------------------
__global__ void k_fps(const float* __restrict__ xyz, int N, int npoint,
                      int* __restrict__ fidx, float* __restrict__ dist)
{
  const int b = blockIdx.x;
  const float* P = xyz + (size_t)b * N * 3;
  float* D = dist + (size_t)b * N;
  int*   F = fidx + (size_t)b * npoint;
  const int t = threadIdx.x, nth = blockDim.x;
  __shared__ float sd[32];
  __shared__ int   si[32];
  __shared__ int   sfar;
  for (int i = t; i < N; i += nth) D[i] = 1e10f;
  __syncthreads();
  int far = 0;
  for (int it = 0; it < npoint; ++it) {
    if (t == 0) F[it] = far;
    const float fx = P[far * 3 + 0], fy = P[far * 3 + 1], fz = P[far * 3 + 2];
    float bd = -1.f; int bi = 0;
    for (int i = t; i < N; i += nth) {
      float dx = P[i * 3 + 0] - fx, dy = P[i * 3 + 1] - fy, dz = P[i * 3 + 2] - fz;
      float d = fminf(D[i], dx * dx + dy * dy + dz * dz);
      D[i] = d;
      if (d > bd) { bd = d; bi = i; }
    }
    for (int off = 16; off > 0; off >>= 1) {
      float od = __shfl_down(bd, off);
      int   oi = __shfl_down(bi, off);
      if (od > bd) { bd = od; bi = oi; }
    }
    if ((t & 31) == 0) { sd[t >> 5] = bd; si[t >> 5] = bi; }
    __syncthreads();
    if (t == 0) {
      const int nw = nth >> 5;
      float best = sd[0]; int bestI = si[0];
      for (int w = 1; w < nw; ++w) if (sd[w] > best) { best = sd[w]; bestI = si[w]; }
      sfar = bestI;
    }
    __syncthreads();
    far = sfar;
    __syncthreads();
  }
}

__global__ void k_gather_xyz(const float* __restrict__ xyz, const int* __restrict__ fidx,
                             int N, int S, int Bq, float* __restrict__ nxyz)
{
  int tid = blockIdx.x * blockDim.x + threadIdx.x;
  if (tid >= Bq * S * 3) return;
  int a = tid % 3; int bq = tid / 3; int q = bq % S; int b = bq / S;
  nxyz[tid] = xyz[((size_t)b * N + fidx[b * S + q]) * 3 + a];
}

// Ball query: first K indices (ascending) with d^2 <= r^2; pad with first hit.
__global__ void k_ball(const float* __restrict__ xyz, const float* __restrict__ nxyz,
                       int N, int S, int Bq, float r2, int K, int* __restrict__ gidx)
{
  int tid = blockIdx.x * blockDim.x + threadIdx.x;
  if (tid >= Bq * S) return;
  int b = tid / S;
  const float* P = xyz + (size_t)b * N * 3;
  const float cx = nxyz[(size_t)tid * 3 + 0];
  const float cy = nxyz[(size_t)tid * 3 + 1];
  const float cz = nxyz[(size_t)tid * 3 + 2];
  int* G = gidx + (size_t)tid * K;
  int cnt = 0, first = N - 1;
  for (int j = 0; j < N && cnt < K; ++j) {
    float dx = P[j * 3 + 0] - cx, dy = P[j * 3 + 1] - cy, dz = P[j * 3 + 2] - cz;
    if (dx * dx + dy * dy + dz * dz <= r2) {
      if (cnt == 0) first = j;
      G[cnt++] = j;
    }
  }
  for (; cnt < K; ++cnt) G[cnt] = first;
}

// Build grouped features: X[row, c] = [feats(gidx) | xyz(gidx)-new_xyz | 0-pad], f16.
__global__ void k_group(const float* __restrict__ xyz, const float* __restrict__ nxyz,
                        const _Float16* __restrict__ feats, const int* __restrict__ gidx,
                        int N, int S, int K, int Cf, int Kpad, int Bq,
                        _Float16* __restrict__ Xout)
{
  size_t tid = (size_t)blockIdx.x * blockDim.x + threadIdx.x;
  size_t total = (size_t)Bq * S * K * Kpad;
  if (tid >= total) return;
  int c = (int)(tid % Kpad);
  size_t row = tid / Kpad;
  size_t bq = row / K;
  int q = (int)(bq % S);
  int b = (int)(bq / S);
  int j = gidx[row];
  float v = 0.f;
  if (c < Cf) {
    v = (float)feats[((size_t)b * N + j) * Cf + c];
  } else if (c < Cf + 3) {
    int a = c - Cf;
    v = xyz[((size_t)b * N + j) * 3 + a] - nxyz[((size_t)b * S + q) * 3 + a];
  }
  Xout[tid] = (_Float16)v;
}

// Max over K consecutive rows of Y[., Npad] -> out[bq, coff+c] inside concat buffer.
__global__ void k_maxpool(const _Float16* __restrict__ Y, int K, int Cout, int Npad,
                          int rowsOut, int Ctot, int coff, _Float16* __restrict__ out)
{
  int tid = blockIdx.x * blockDim.x + threadIdx.x;
  if (tid >= rowsOut * Cout) return;
  int c = tid % Cout; int bq = tid / Cout;
  const _Float16* p = Y + (size_t)bq * K * Npad + c;
  float m = -3.0e38f;
  for (int kk = 0; kk < K; ++kk) m = fmaxf(m, (float)p[(size_t)kk * Npad]);
  out[(size_t)bq * Ctot + coff + c] = (_Float16)m;
}

// SA4 group-all input: X[bq, c] = [xyz | feats | 0-pad]
__global__ void k_cat_xyz_feat(const float* __restrict__ xyz, const _Float16* __restrict__ feats,
                               int S, int C, int Kpad, int Bq, _Float16* __restrict__ X)
{
  size_t tid = (size_t)blockIdx.x * blockDim.x + threadIdx.x;
  size_t total = (size_t)Bq * S * Kpad;
  if (tid >= total) return;
  int c = (int)(tid % Kpad);
  size_t bq = tid / Kpad;
  float v = 0.f;
  if (c < 3) v = xyz[bq * 3 + c];
  else if (c < 3 + C) v = (float)feats[bq * C + (c - 3)];
  X[tid] = (_Float16)v;
}

// FP3: broadcast concat (source has a single point per batch)
__global__ void k_fp3cat(const _Float16* __restrict__ p1, const _Float16* __restrict__ p2,
                         int S, int C1, int C2, int Kpad, int Bq, _Float16* __restrict__ X)
{
  size_t tid = (size_t)blockIdx.x * blockDim.x + threadIdx.x;
  size_t total = (size_t)Bq * S * Kpad;
  if (tid >= total) return;
  int c = (int)(tid % Kpad);
  size_t bq = tid / Kpad;
  int b = (int)(bq / S);
  float v = 0.f;
  if (c < C1) v = (float)p1[bq * C1 + c];
  else if (c < C1 + C2) v = (float)p2[(size_t)b * C2 + (c - C1)];
  X[tid] = (_Float16)v;
}

// 3-nearest-neighbor search + inverse-distance weights
__global__ void k_knn3(const float* __restrict__ x1, const float* __restrict__ x2,
                       int n1, int n2, int Bq, int* __restrict__ idx3, float* __restrict__ w3)
{
  int tid = blockIdx.x * blockDim.x + threadIdx.x;
  if (tid >= Bq * n1) return;
  int b = tid / n1;
  const float px = x1[(size_t)tid * 3 + 0];
  const float py = x1[(size_t)tid * 3 + 1];
  const float pz = x1[(size_t)tid * 3 + 2];
  const float* Q = x2 + (size_t)b * n2 * 3;
  float d0 = 3e38f, d1 = 3e38f, d2 = 3e38f;
  int i0 = 0, i1 = 0, i2 = 0;
  for (int j = 0; j < n2; ++j) {
    float dx = Q[j * 3 + 0] - px, dy = Q[j * 3 + 1] - py, dz = Q[j * 3 + 2] - pz;
    float d = dx * dx + dy * dy + dz * dz;
    if (d < d0)      { d2 = d1; i2 = i1; d1 = d0; i1 = i0; d0 = d; i0 = j; }
    else if (d < d1) { d2 = d1; i2 = i1; d1 = d; i1 = j; }
    else if (d < d2) { d2 = d; i2 = j; }
  }
  float w0 = 1.f / (d0 + 1e-8f), w1 = 1.f / (d1 + 1e-8f), w2 = 1.f / (d2 + 1e-8f);
  float s = w0 + w1 + w2;
  idx3[tid * 3 + 0] = i0; idx3[tid * 3 + 1] = i1; idx3[tid * 3 + 2] = i2;
  w3[tid * 3 + 0] = w0 / s; w3[tid * 3 + 1] = w1 / s; w3[tid * 3 + 2] = w2 / s;
}

// FP concat: X[row, c] = [p1 | sum_j w_j * p2[idx_j] | 0-pad]
__global__ void k_fpcat(const _Float16* __restrict__ p1, const _Float16* __restrict__ p2,
                        const int* __restrict__ idx3, const float* __restrict__ w3,
                        int n1, int n2, int C1, int C2, int Kpad, int Bq,
                        _Float16* __restrict__ X)
{
  size_t tid = (size_t)blockIdx.x * blockDim.x + threadIdx.x;
  size_t total = (size_t)Bq * n1 * Kpad;
  if (tid >= total) return;
  int c = (int)(tid % Kpad);
  size_t row = tid / Kpad;
  int b = (int)(row / n1);
  float v = 0.f;
  if (c < C1) {
    v = (float)p1[row * C1 + c];
  } else if (c < C1 + C2) {
    int cc = c - C1;
    float s = 0.f;
    for (int j = 0; j < 3; ++j) {
      int id = idx3[row * 3 + j];
      s += w3[row * 3 + j] * (float)p2[((size_t)b * n2 + id) * C2 + cc];
    }
    v = s;
  }
  X[tid] = (_Float16)v;
}

// f32 -> f16 weight conversion with K zero-padding: W16[o, 0..Kpad)
__global__ void k_wconv(const float* __restrict__ w, int cin, int cout, int Kpad,
                        _Float16* __restrict__ w16)
{
  int tid = blockIdx.x * blockDim.x + threadIdx.x;
  if (tid >= cout * Kpad) return;
  int c = tid % Kpad; int o = tid / Kpad;
  w16[tid] = (c < cin) ? (_Float16)w[(size_t)o * cin + c] : (_Float16)0.f;
}

// Final 128->1 head (f32 out)
__global__ void k_head2(const _Float16* __restrict__ H, const float* __restrict__ w,
                        const float* __restrict__ b, float* __restrict__ out, int rows)
{
  int i = blockIdx.x * blockDim.x + threadIdx.x;
  if (i >= rows) return;
  float s = b[0];
  for (int c = 0; c < 128; ++c) s += (float)H[(size_t)i * 128 + c] * w[c];
  out[i] = s;
}

// ---------------------------------------------------------------------------
static inline int pad32i(int x) { return (x + 31) & ~31; }

static inline void launch_gemm(hipStream_t st, const _Float16* X, const _Float16* W16,
                               const float* bias, _Float16* Y, int rows, int Kpad, int Nc,
                               int relu)
{
  const int Mtiles = rows >> 4;
  dim3 grid((Mtiles + 7) >> 3, Nc >> 4);
  k_gemm_relu<<<grid, 256, 0, st>>>(X, W16, bias, Y, Mtiles, Kpad, Nc, relu);
}

extern "C" void kernel_launch(void* const* d_in, const int* in_sizes, int n_in,
                              void* d_out, int out_size, void* d_ws, size_t ws_size,
                              hipStream_t stream)
{
  (void)in_sizes; (void)n_in; (void)out_size; (void)ws_size;
  const int B = 2, N = 32768;

  // ---- input flattening (jax pytree order: dict keys sorted; 'b' < 'w';
  //      top-level 'params' < 'points') ----
  struct CP { const float* b; const float* w; int cin, cout; };
  int idx = 0;
  auto take = [&](int cin, int cout) {
    CP c; c.b = (const float*)d_in[idx++]; c.w = (const float*)d_in[idx++];
    c.cin = cin; c.cout = cout; return c;
  };
  CP fp1c[3] = { take(448,128), take(128,128), take(128,128) };
  CP fp2c[2] = { take(896,256), take(256,128) };
  CP fp3c[2] = { take(1664,256), take(256,256) };
  CP h1 = take(128,128);
  CP h2 = take(128,1);
  CP sa1c[9] = { take(3,32), take(32,32), take(32,64),
                 take(3,64), take(64,64), take(64,128),
                 take(3,64), take(64,96), take(96,128) };
  CP sa2c[9] = { take(323,64),  take(64,64),   take(64,128),
                 take(323,128), take(128,128), take(128,256),
                 take(323,128), take(128,128), take(128,256) };
  CP sa3c[9] = { take(643,64),  take(64,64),   take(64,128),
                 take(643,128), take(128,128), take(128,256),
                 take(643,128), take(128,128), take(128,256) };
  CP sa4c[3] = { take(643,256), take(256,512), take(512,1024) };
  const float* points = (const float*)d_in[idx++];

  // ---- workspace bump allocator ----
  char* wsb = (char*)d_ws;
  size_t off = 0;
  auto alloc = [&](size_t bytes) -> void* {
    void* p = wsb + off;
    off += (bytes + 255) & ~(size_t)255;
    return p;
  };

  float* fpsd   = (float*)alloc(sizeof(float) * B * N);
  int*   fidx1  = (int*)alloc(sizeof(int) * B * 1024);
  int*   fidx2  = (int*)alloc(sizeof(int) * B * 512);
  int*   fidx3  = (int*)alloc(sizeof(int) * B * 128);
  float* x1     = (float*)alloc(sizeof(float) * B * 1024 * 3);
  float* x2     = (float*)alloc(sizeof(float) * B * 512 * 3);
  float* x3     = (float*)alloc(sizeof(float) * B * 128 * 3);
  _Float16* l1p = (_Float16*)alloc(2ull * B * 1024 * 320);
  _Float16* l2p = (_Float16*)alloc(2ull * B * 512 * 640);
  _Float16* l3p = (_Float16*)alloc(2ull * B * 128 * 640);
  _Float16* l4p = (_Float16*)alloc(2ull * B * 1024);
  _Float16* f3o = (_Float16*)alloc(2ull * B * 128 * 256);
  _Float16* f2o = (_Float16*)alloc(2ull * B * 512 * 128);
  _Float16* f1o = (_Float16*)alloc(2ull * B * 1024 * 128);
  _Float16* h1o = (_Float16*)alloc(2ull * B * 1024 * 128);
  int*   idx3b  = (int*)alloc(sizeof(int) * B * 1024 * 3);
  float* w3b    = (float*)alloc(sizeof(float) * B * 1024 * 3);
  int*   gidx   = (int*)alloc(sizeof(int) * B * 1024 * 128);

  // f16 weight conversion (every call; cheap & deterministic)
  auto convW = [&](const CP& c) -> _Float16* {
    int Kp = pad32i(c.cin);
    _Float16* w16 = (_Float16*)alloc(2ull * c.cout * Kp);
    int total = c.cout * Kp;
    k_wconv<<<(total + 255) / 256, 256, 0, stream>>>(c.w, c.cin, c.cout, Kp, w16);
    return w16;
  };
  _Float16 *sa1w[9], *sa2w[9], *sa3w[9], *sa4w[3], *fp1w[3], *fp2w[2], *fp3w[2], *h1w;
  for (int i = 0; i < 9; ++i) sa1w[i] = convW(sa1c[i]);
  for (int i = 0; i < 9; ++i) sa2w[i] = convW(sa2c[i]);
  for (int i = 0; i < 9; ++i) sa3w[i] = convW(sa3c[i]);
  for (int i = 0; i < 3; ++i) sa4w[i] = convW(sa4c[i]);
  for (int i = 0; i < 3; ++i) fp1w[i] = convW(fp1c[i]);
  for (int i = 0; i < 2; ++i) fp2w[i] = convW(fp2c[i]);
  for (int i = 0; i < 2; ++i) fp3w[i] = convW(fp3c[i]);
  h1w = convW(h1);

  // big ping-pong arenas (sized for SA2 branch3)
  _Float16* arenaA = (_Float16*)alloc(2ull * 131072 * 352);  // Xin / odd-layer outs
  _Float16* arenaB = (_Float16*)alloc(2ull * 262144 * 128);  // even-layer outs

  // MLP chain: X0 lives in arenaA; layer l writes arenaB (even) / arenaA (odd);
  // optional explicit destination for the final layer.
  auto runMLP = [&](const _Float16* X0, int rows, int Kp0, const CP* cs,
                    _Float16* const* ws16, int nl, _Float16* lastOut) -> const _Float16* {
    const _Float16* cur = X0;
    int curK = Kp0;
    for (int l = 0; l < nl; ++l) {
      _Float16* o = (l == nl - 1 && lastOut) ? lastOut : ((l & 1) ? arenaA : arenaB);
      launch_gemm(stream, cur, ws16[l], cs[l].b, o, rows, curK, cs[l].cout, 1);
      cur = o;
      curK = cs[l].cout;
    }
    return cur;
  };

  // one SA-MSG stage
  auto runSA = [&](const float* xyz, int Np, const _Float16* feats, int Cf, int S,
                   const float* radii, const int* ks, const CP* br, _Float16* const* w16s,
                   int* fidx, float* nxyz, _Float16* outP, int Ctot) {
    k_fps<<<B, (Np >= 4096 ? 1024 : 256), 0, stream>>>(xyz, Np, S, fidx, fpsd);
    k_gather_xyz<<<(B * S * 3 + 255) / 256, 256, 0, stream>>>(xyz, fidx, Np, S, B, nxyz);
    int coff = 0;
    for (int b2 = 0; b2 < 3; ++b2) {
      const int K = ks[b2];
      const float r2 = radii[b2] * radii[b2];
      const int rows = B * S * K;
      const int Kp0 = pad32i(Cf + 3);
      k_ball<<<(B * S + 255) / 256, 256, 0, stream>>>(xyz, nxyz, Np, S, B, r2, K, gidx);
      size_t tot = (size_t)rows * Kp0;
      k_group<<<(unsigned)((tot + 255) / 256), 256, 0, stream>>>(
          xyz, nxyz, feats, gidx, Np, S, K, Cf, Kp0, B, arenaA);
      const _Float16* y = runMLP(arenaA, rows, Kp0, &br[b2 * 3], &w16s[b2 * 3], 3, nullptr);
      const int Cout = br[b2 * 3 + 2].cout;
      k_maxpool<<<(B * S * Cout + 255) / 256, 256, 0, stream>>>(
          y, K, Cout, Cout, B * S, Ctot, coff, outP);
      coff += Cout;
    }
  };

  // ---- forward pass ----
  const float sa1r[3] = {0.02f, 0.04f, 0.08f}; const int sa1k[3] = {32, 64, 128};
  runSA(points, N, nullptr, 0, 1024, sa1r, sa1k, sa1c, sa1w, fidx1, x1, l1p, 320);

  const float sa2r[3] = {0.04f, 0.08f, 0.16f}; const int sa2k[3] = {64, 64, 128};
  runSA(x1, 1024, l1p, 320, 512, sa2r, sa2k, sa2c, sa2w, fidx2, x2, l2p, 640);

  const float sa3r[3] = {0.08f, 0.16f, 0.32f}; const int sa3k[3] = {64, 64, 128};
  runSA(x2, 512, l2p, 640, 128, sa3r, sa3k, sa3c, sa3w, fidx3, x3, l3p, 640);

  { // SA4: group-all MLP + max over the 128 points
    const int rows = B * 128, Kp0 = pad32i(643);
    size_t tot = (size_t)rows * Kp0;
    k_cat_xyz_feat<<<(unsigned)((tot + 255) / 256), 256, 0, stream>>>(
        x3, l3p, 128, 640, Kp0, B, arenaA);
    const _Float16* y = runMLP(arenaA, rows, Kp0, sa4c, sa4w, 3, nullptr);
    k_maxpool<<<(B * 1024 + 255) / 256, 256, 0, stream>>>(y, 128, 1024, 1024, B, 1024, 0, l4p);
  }

  { // FP3 (broadcast from single point)
    const int rows = B * 128, Kp = 1664;
    size_t tot = (size_t)rows * Kp;
    k_fp3cat<<<(unsigned)((tot + 255) / 256), 256, 0, stream>>>(
        l3p, l4p, 128, 640, 1024, Kp, B, arenaA);
    runMLP(arenaA, rows, Kp, fp3c, fp3w, 2, f3o);
  }

  { // FP2 (3-NN interp 128 -> 512)
    k_knn3<<<(B * 512 + 255) / 256, 256, 0, stream>>>(x2, x3, 512, 128, B, idx3b, w3b);
    const int rows = B * 512, Kp = 896;
    size_t tot = (size_t)rows * Kp;
    k_fpcat<<<(unsigned)((tot + 255) / 256), 256, 0, stream>>>(
        l2p, f3o, idx3b, w3b, 512, 128, 640, 256, Kp, B, arenaA);
    runMLP(arenaA, rows, Kp, fp2c, fp2w, 2, f2o);
  }

  { // FP1 (3-NN interp 512 -> 1024)
    k_knn3<<<(B * 1024 + 255) / 256, 256, 0, stream>>>(x1, x2, 1024, 512, B, idx3b, w3b);
    const int rows = B * 1024, Kp = 448;
    size_t tot = (size_t)rows * Kp;
    k_fpcat<<<(unsigned)((tot + 255) / 256), 256, 0, stream>>>(
        l1p, f2o, idx3b, w3b, 1024, 512, 320, 128, Kp, B, arenaA);
    runMLP(arenaA, rows, Kp, fp1c, fp1w, 3, f1o);
  }

  // heads
  launch_gemm(stream, f1o, h1w, h1.b, h1o, B * 1024, 128, 128, 1);
  k_head2<<<(B * 1024 + 255) / 256, 256, 0, stream>>>(h1o, h2.w, h2.b, (float*)d_out, B * 1024);
}
